// MHA_17849884082233
// MI455X (gfx1250) — matrix operand order repaired and checked by reading, not compiled, not run on hardware
//
#include <hip/hip_runtime.h>
#include <hip/hip_bf16.h>

// ---------------------------------------------------------------------------
// Fused MHA block for MI455X (gfx1250, wave32, WMMA f32_16x16x32_f16)
//   out = [ y (B,S,D) | k_rot (B,H,S,HD) | v (B,H,S,HD) ]
// ---------------------------------------------------------------------------

typedef _Float16 v16h __attribute__((ext_vector_type(16)));
typedef _Float16 v8h  __attribute__((ext_vector_type(8)));
typedef _Float16 v4h  __attribute__((ext_vector_type(4)));
typedef float    v8f  __attribute__((ext_vector_type(8)));

#define WMMA_F16(a, b, c) \
    __builtin_amdgcn_wmma_f32_16x16x32_f16(false, (a), false, (b), (short)0, (c), false, false)

constexpr int Bb  = 2;
constexpr int Ss  = 2048;
constexpr int Dd  = 1024;
constexpr int Hh  = 16;
constexpr int HD  = 64;
constexpr int BS  = Bb * Ss;            // 4096 tokens
constexpr long BSD = (long)BS * Dd;     // 4,194,304
constexpr int LDH = 72;                 // f16 LDS row stride (144B, 16B aligned)
constexpr int KT  = 64;                 // GEMM K tile

// A-matrix fragment (16x32 f16): lane L holds M=L&15; K=kbase+(L>>4)*16+e.
__device__ inline v16h frag_a(const _Float16* s, int rbase, int kbase, int lane) {
    const _Float16* p = s + (rbase + (lane & 15)) * LDH + kbase + ((lane >> 4) << 4);
    v8h lo = *(const v8h*)p;
    v8h hi = *(const v8h*)(p + 8);
    v16h a;
#pragma unroll
    for (int i = 0; i < 8; ++i) { a[i] = lo[i]; a[8 + i] = hi[i]; }
    return a;
}

// B-matrix fragment (32x16 f16): lane L holds N=nbase+(L&15); K=kbase+(L>>4)*16+e.
// Source LDS tile stored N-major [N][K] so the 16 halves are contiguous.
__device__ inline v16h frag_b(const _Float16* s, int nbase, int kbase, int lane) {
    const _Float16* p = s + (nbase + (lane & 15)) * LDH + kbase + ((lane >> 4) << 4);
    return *(const v16h*)p;
}

// 64x64 MACC step (attention): load all fragments, then 8 WMMAs back-to-back.
__device__ inline void macc_64x64(const _Float16* As, const _Float16* BsT,
                                  int w, int lane, v8f acc[4]) {
    v16h a0 = frag_a(As, w * 16, 0, lane);
    v16h a1 = frag_a(As, w * 16, 32, lane);
    v16h b00 = frag_b(BsT, 0, 0, lane),  b01 = frag_b(BsT, 16, 0, lane);
    v16h b02 = frag_b(BsT, 32, 0, lane), b03 = frag_b(BsT, 48, 0, lane);
    v16h b10 = frag_b(BsT, 0, 32, lane),  b11 = frag_b(BsT, 16, 32, lane);
    v16h b12 = frag_b(BsT, 32, 32, lane), b13 = frag_b(BsT, 48, 32, lane);
    acc[0] = WMMA_F16(a0, b00, acc[0]);
    acc[1] = WMMA_F16(a0, b01, acc[1]);
    acc[2] = WMMA_F16(a0, b02, acc[2]);
    acc[3] = WMMA_F16(a0, b03, acc[3]);
    acc[0] = WMMA_F16(a1, b10, acc[0]);
    acc[1] = WMMA_F16(a1, b11, acc[1]);
    acc[2] = WMMA_F16(a1, b12, acc[2]);
    acc[3] = WMMA_F16(a1, b13, acc[3]);
}

// 128x64 MACC step (GEMMs): each wave owns a 32-row strip; B fragments are
// reused across the two 16-row sub-strips (16 WMMAs per 24 ds_load_b128).
__device__ inline void macc_128x64(const _Float16* As, const _Float16* BsT,
                                   int w, int lane, v8f acc0[4], v8f acc1[4]) {
#pragma unroll
    for (int ks = 0; ks < 2; ++ks) {
        v16h a0 = frag_a(As, w * 32,      ks * 32, lane);
        v16h a1 = frag_a(As, w * 32 + 16, ks * 32, lane);
        v16h b0 = frag_b(BsT, 0,  ks * 32, lane);
        v16h b1 = frag_b(BsT, 16, ks * 32, lane);
        v16h b2 = frag_b(BsT, 32, ks * 32, lane);
        v16h b3 = frag_b(BsT, 48, ks * 32, lane);
        acc0[0] = WMMA_F16(a0, b0, acc0[0]);
        acc0[1] = WMMA_F16(a0, b1, acc0[1]);
        acc0[2] = WMMA_F16(a0, b2, acc0[2]);
        acc0[3] = WMMA_F16(a0, b3, acc0[3]);
        acc1[0] = WMMA_F16(a1, b0, acc1[0]);
        acc1[1] = WMMA_F16(a1, b1, acc1[1]);
        acc1[2] = WMMA_F16(a1, b2, acc1[2]);
        acc1[3] = WMMA_F16(a1, b3, acc1[3]);
    }
}

// Stage a 64-row x 64-col f32 block (rows rb0..rb0+63 of src) into f16 LDS
// [row][k], batched: 8 loads in flight, then 8 stores.
__device__ inline void stage_a64(const float* src, size_t ldsrc, int rb0, int kcol,
                                 _Float16* dstLds, int t) {
    float4 f[8];
#pragma unroll
    for (int i = 0; i < 8; ++i) {
        int idx = i * 128 + t;
        int r = idx >> 4, k = (idx & 15) << 2;
        f[i] = *(const float4*)(src + (size_t)(rb0 + r) * ldsrc + kcol + k);
    }
#pragma unroll
    for (int i = 0; i < 8; ++i) {
        int idx = i * 128 + t;
        int r = idx >> 4, k = (idx & 15) << 2;
        v4h hv = {(_Float16)f[i].x, (_Float16)f[i].y,
                  (_Float16)f[i].z, (_Float16)f[i].w};
        *(v4h*)&dstLds[r * LDH + k] = hv;
    }
}

// Stage a 64(K) x 64(N) f32 weight block into transposed f16 LDS [n][k].
__device__ inline void stage_bT(const float* W, size_t ldw, int k0, int colBase,
                                _Float16* dstLds, int t) {
    float4 f[8];
#pragma unroll
    for (int i = 0; i < 8; ++i) {
        int idx = i * 128 + t;
        int kr = idx >> 4, n = (idx & 15) << 2;
        f[i] = *(const float4*)(W + (size_t)(k0 + kr) * ldw + colBase + n);
    }
#pragma unroll
    for (int i = 0; i < 8; ++i) {
        int idx = i * 128 + t;
        int kr = idx >> 4, n = (idx & 15) << 2;
        dstLds[(n + 0) * LDH + kr] = (_Float16)f[i].x;
        dstLds[(n + 1) * LDH + kr] = (_Float16)f[i].y;
        dstLds[(n + 2) * LDH + kr] = (_Float16)f[i].z;
        dstLds[(n + 3) * LDH + kr] = (_Float16)f[i].w;
    }
}

// ---------------------------------------------------------------------------
// Kernel 1: qkv = x @ Wqkv + bqkv, RoPE on q and k, scatter to head layout.
//   q (rotated)  -> out[0 : BSD)           (staging; overwritten by proj later)
//   k (rotated)  -> out[BSD : 2*BSD)
//   v            -> out[2*BSD : 3*BSD)
// grid = (BS/128, 48), block = 128 (4 waves); each wave: 32 rows x 64 cols.
// ---------------------------------------------------------------------------
__global__ __launch_bounds__(128) void qkv_rope_kernel(
    const float* __restrict__ x, const float* __restrict__ cosb,
    const float* __restrict__ sinb, const float* __restrict__ W,
    const float* __restrict__ bias, float* __restrict__ out)
{
    __shared__ alignas(16) _Float16 As[128 * LDH];  // x tile, row-major [row][k]
    __shared__ alignas(16) _Float16 BsT[64 * LDH];  // W tile, transposed [n][k]

    const int t = threadIdx.x, lane = t & 31, w = t >> 5;
    const int rowBase = blockIdx.x * 128;           // token tile
    const int ct      = blockIdx.y;                 // col tile (0..47)
    const int seg     = ct >> 4;                    // 0=q 1=k 2=v
    const int h       = ct & 15;
    const int colBase = ct * 64;

    v8f acc0[4], acc1[4];
#pragma unroll
    for (int nt = 0; nt < 4; ++nt)
#pragma unroll
        for (int r = 0; r < 8; ++r) { acc0[nt][r] = 0.0f; acc1[nt][r] = 0.0f; }

    for (int k0 = 0; k0 < Dd; k0 += KT) {
        __syncthreads();
        stage_a64(x, Dd, rowBase,      k0, As,            t);
        stage_a64(x, Dd, rowBase + 64, k0, As + 64 * LDH, t);
        stage_bT(W, 3 * Dd, k0, colBase, BsT, t);
        __syncthreads();
        macc_128x64(As, BsT, w, lane, acc0, acc1);
    }

    // Epilogue: bias, RoPE (q/k), scatter to (B,H,S,HD).
    const int b = rowBase >> 11;
    float* dst = out + (size_t)seg * BSD + ((size_t)(b * Hh + h)) * Ss * HD;
#pragma unroll
    for (int strip = 0; strip < 2; ++strip) {
#pragma unroll
        for (int nt = 0; nt < 4; ++nt) {
            const int hd = nt * 16 + (lane & 15);
            const float bv = bias[colBase + hd];
#pragma unroll
            for (int r = 0; r < 8; ++r) {
                const int s = (rowBase & 2047) + w * 32 + strip * 16 + r +
                              ((lane >> 4) << 3);
                float v = (strip == 0 ? acc0[nt][r] : acc1[nt][r]) + bv;
                if (seg == 2) {
                    dst[(size_t)s * HD + hd] = v;
                } else {
                    float partner = __shfl_xor(v, 1, 32);  // adjacent-hd pair
                    float c = cosb[s * HD + hd], sn = sinb[s * HD + hd];
                    float o = ((hd & 1) == 0) ? (v * c - partner * sn)
                                              : (v * c + partner * sn);
                    dst[(size_t)s * HD + hd] = o;
                }
            }
        }
    }
}

// ---------------------------------------------------------------------------
// Kernel 2: flash attention per (b,h, 64-row q tile). Reads q/k/v from d_out,
// writes y (B,S,D) into workspace. grid = B*H*(S/64) = 1024, block = 128.
// ---------------------------------------------------------------------------
__global__ __launch_bounds__(128) void attn_kernel(
    const float* __restrict__ qkv, float* __restrict__ y)
{
    __shared__ alignas(16) _Float16 Qs[64 * LDH];   // [qrow][hd]
    __shared__ alignas(16) _Float16 Ks[64 * LDH];   // [key][hd]  (B for Q@K^T)
    __shared__ alignas(16) _Float16 VsT[64 * LDH];  // [hd][key]  (B for P@V)
    __shared__ alignas(16) _Float16 Ps[64 * LDH];   // [qrow][key]
    __shared__ alignas(16) float Sc[64 * 66];       // raw scores f32
    __shared__ float mrow[64], lrow[64], arow[64];

    const int t = threadIdx.x, lane = t & 31, w = t >> 5;
    const int qt = blockIdx.x & 31;
    const int bh = blockIdx.x >> 5;
    const float* qb = qkv + (size_t)bh * Ss * HD;
    const float* kb = qkv + (size_t)BSD + (size_t)bh * Ss * HD;
    const float* vb = qkv + (size_t)2 * BSD + (size_t)bh * Ss * HD;

    const int lr = t >> 4, ld0 = (t & 15) << 2;     // staging coords
#pragma unroll
    for (int i = 0; i < 8; ++i) {                   // Q tile once (rows 0..63)
        int r = lr + i * 8;
        float4 f = *(const float4*)(qb + (size_t)(qt * 64 + r) * HD + ld0);
        v4h hv = {(_Float16)f.x, (_Float16)f.y, (_Float16)f.z, (_Float16)f.w};
        *(v4h*)&Qs[r * LDH + ld0] = hv;
    }
    if (t < 64) { mrow[t] = -3.0e38f; lrow[t] = 0.0f; }

    v8f o[4];
#pragma unroll
    for (int nt = 0; nt < 4; ++nt)
#pragma unroll
        for (int r = 0; r < 8; ++r) o[nt][r] = 0.0f;
    __syncthreads();

    for (int jt = 0; jt < Ss / 64; ++jt) {
        if (jt + 1 < Ss / 64) {                     // -> global_prefetch_b8
            __builtin_prefetch(kb + (size_t)(jt + 1) * 64 * HD, 0, 0);
            __builtin_prefetch(vb + (size_t)(jt + 1) * 64 * HD, 0, 0);
        }
        float4 fk[4], fv[4];
#pragma unroll
        for (int i = 0; i < 4; ++i) {               // 8 loads in flight
            int r = lr + i * 16;                    // rows 0-7,16-23,32-39,48-55
            fk[i] = *(const float4*)(kb + (size_t)(jt * 64 + r) * HD + ld0);
            fv[i] = *(const float4*)(vb + (size_t)(jt * 64 + r + 8) * HD + ld0);
        }
#pragma unroll
        for (int i = 0; i < 4; ++i) {               // K row-major, V transposed
            int r = lr + i * 16;
            v4h hv = {(_Float16)fk[i].x, (_Float16)fk[i].y,
                      (_Float16)fk[i].z, (_Float16)fk[i].w};
            *(v4h*)&Ks[r * LDH + ld0] = hv;
            VsT[(ld0 + 0) * LDH + r + 8] = (_Float16)fv[i].x;
            VsT[(ld0 + 1) * LDH + r + 8] = (_Float16)fv[i].y;
            VsT[(ld0 + 2) * LDH + r + 8] = (_Float16)fv[i].z;
            VsT[(ld0 + 3) * LDH + r + 8] = (_Float16)fv[i].w;
        }
        float4 gk[4], gv[4];
#pragma unroll
        for (int i = 0; i < 4; ++i) {               // remaining rows (roles swap)
            int r = lr + i * 16;                    // rows 8-15,24-31,40-47,56-63
            gk[i] = *(const float4*)(kb + (size_t)(jt * 64 + r + 8) * HD + ld0);
            gv[i] = *(const float4*)(vb + (size_t)(jt * 64 + r) * HD + ld0);
        }
#pragma unroll
        for (int i = 0; i < 4; ++i) {
            int r = lr + i * 16;
            v4h hv = {(_Float16)gk[i].x, (_Float16)gk[i].y,
                      (_Float16)gk[i].z, (_Float16)gk[i].w};
            *(v4h*)&Ks[(r + 8) * LDH + ld0] = hv;
            VsT[(ld0 + 0) * LDH + r] = (_Float16)gv[i].x;
            VsT[(ld0 + 1) * LDH + r] = (_Float16)gv[i].y;
            VsT[(ld0 + 2) * LDH + r] = (_Float16)gv[i].z;
            VsT[(ld0 + 3) * LDH + r] = (_Float16)gv[i].w;
        }
        __syncthreads();

        v8f sc[4];                                  // S = Q @ K^T (contract hd)
#pragma unroll
        for (int nt = 0; nt < 4; ++nt)
#pragma unroll
            for (int r = 0; r < 8; ++r) sc[nt][r] = 0.0f;
        macc_64x64(Qs, Ks, w, lane, sc);
#pragma unroll
        for (int nt = 0; nt < 4; ++nt)
#pragma unroll
            for (int r = 0; r < 8; ++r)
                Sc[(w * 16 + r + ((lane >> 4) << 3)) * 66 + nt * 16 + (lane & 15)] =
                    sc[nt][r] * 0.125f;             // 1/sqrt(64)
        __syncthreads();

        if (t < 64) {                               // online softmax, row t
            const float* srow = &Sc[t * 66];
            float m_old = mrow[t], mx = m_old;
            for (int j = 0; j < 64; ++j) mx = fmaxf(mx, srow[j]);
            float al = __expf(m_old - mx), sum = 0.0f;
            _Float16* prow = &Ps[t * LDH];
            for (int j = 0; j < 64; ++j) {
                float p = __expf(srow[j] - mx);
                sum += p;
                prow[j] = (_Float16)p;
            }
            lrow[t] = lrow[t] * al + sum;
            mrow[t] = mx;
            arow[t] = al;
        }
        __syncthreads();

        float alr[8];                               // rescale O accumulators
#pragma unroll
        for (int r = 0; r < 8; ++r) alr[r] = arow[w * 16 + r + ((lane >> 4) << 3)];
#pragma unroll
        for (int nt = 0; nt < 4; ++nt)
#pragma unroll
            for (int r = 0; r < 8; ++r) o[nt][r] *= alr[r];

        macc_64x64(Ps, VsT, w, lane, o);            // O += P @ V (contract key)
        __syncthreads();
    }

    float linv[8];
#pragma unroll
    for (int r = 0; r < 8; ++r)
        linv[r] = 1.0f / lrow[w * 16 + r + ((lane >> 4) << 3)];
    const int b = bh >> 4, h = bh & 15;
    float* yb = y + (size_t)b * Ss * Dd + (size_t)h * HD;
#pragma unroll
    for (int nt = 0; nt < 4; ++nt) {
        const int hd = nt * 16 + (lane & 15);
#pragma unroll
        for (int r = 0; r < 8; ++r) {
            const int s = qt * 64 + w * 16 + r + ((lane >> 4) << 3);
            yb[(size_t)s * Dd + hd] = o[nt][r] * linv[r];
        }
    }
}

// ---------------------------------------------------------------------------
// Kernel 3: y = yin @ Wproj + bproj  -> out[0 : BSD), (BS x Dd) row-major.
// grid = (BS/128, Dd/64), block = 128 (4 waves); each wave: 32 rows x 64 cols.
// ---------------------------------------------------------------------------
__global__ __launch_bounds__(128) void proj_kernel(
    const float* __restrict__ yin, const float* __restrict__ W,
    const float* __restrict__ bias, float* __restrict__ out)
{
    __shared__ alignas(16) _Float16 As[128 * LDH];
    __shared__ alignas(16) _Float16 BsT[64 * LDH];

    const int t = threadIdx.x, lane = t & 31, w = t >> 5;
    const int rowBase = blockIdx.x * 128;
    const int colBase = blockIdx.y * 64;

    v8f acc0[4], acc1[4];
#pragma unroll
    for (int nt = 0; nt < 4; ++nt)
#pragma unroll
        for (int r = 0; r < 8; ++r) { acc0[nt][r] = 0.0f; acc1[nt][r] = 0.0f; }

    for (int k0 = 0; k0 < Dd; k0 += KT) {
        __syncthreads();
        stage_a64(yin, Dd, rowBase,      k0, As,            t);
        stage_a64(yin, Dd, rowBase + 64, k0, As + 64 * LDH, t);
        stage_bT(W, Dd, k0, colBase, BsT, t);
        __syncthreads();
        macc_128x64(As, BsT, w, lane, acc0, acc1);
    }

#pragma unroll
    for (int strip = 0; strip < 2; ++strip) {
#pragma unroll
        for (int nt = 0; nt < 4; ++nt) {
            const int c = colBase + nt * 16 + (lane & 15);
            const float bv = bias[c];
#pragma unroll
            for (int r = 0; r < 8; ++r) {
                const int g = rowBase + w * 32 + strip * 16 + r + ((lane >> 4) << 3);
                out[(size_t)g * Dd + c] =
                    (strip == 0 ? acc0[nt][r] : acc1[nt][r]) + bv;
            }
        }
    }
}

extern "C" void kernel_launch(void* const* d_in, const int* in_sizes, int n_in,
                              void* d_out, int out_size, void* d_ws, size_t ws_size,
                              hipStream_t stream)
{
    (void)in_sizes; (void)n_in; (void)out_size; (void)ws_size;
    const float* x     = (const float*)d_in[0];
    const float* cosb  = (const float*)d_in[1];
    const float* sinb  = (const float*)d_in[2];
    const float* Wqkv  = (const float*)d_in[3];
    const float* bqkv  = (const float*)d_in[4];
    const float* Wproj = (const float*)d_in[5];
    const float* bproj = (const float*)d_in[6];
    float* out = (float*)d_out;
    float* yws = (float*)d_ws;  // BS * Dd floats (16 MB)

    qkv_rope_kernel<<<dim3(BS / 128, 48), 128, 0, stream>>>(x, cosb, sinb, Wqkv, bqkv, out);
    attn_kernel<<<dim3(Bb * Hh * (Ss / 64)), 128, 0, stream>>>(out, yws);
    proj_kernel<<<dim3(BS / 128, Dd / 64), 128, 0, stream>>>(yws, Wproj, bproj, out);
}
